// StructuralEncoderLayer_41180146434894
// MI455X (gfx1250) — 8-bit matrix operand order repaired and checked by reading, compiled, not run on hardware
//
#include <hip/hip_runtime.h>
#include <stdint.h>


#define Nn   4096
#define Ee   131072
#define TILE 128
#define FP8_ONE ((uint8_t)0x38)

typedef __attribute__((ext_vector_type(16))) int   v16i;
typedef __attribute__((ext_vector_type(8)))  float v8f;

union AFrag { v16i v; uint2 d[8]; };
union BFrag { v16i v; uint4 q[4]; };

__device__ __forceinline__ void async_ld16(uint32_t lds_off, const uint8_t* gaddr) {
    asm volatile("global_load_async_to_lds_b128 %0, %1, off"
                 :: "v"(lds_off), "v"(gaddr) : "memory");
}
__device__ __forceinline__ void wait_async_zero() {
    asm volatile("s_wait_asynccnt 0x0" ::: "memory");
}

__global__ void k_zero(uint4* __restrict__ p, int n16) {
    int i = blockIdx.x * blockDim.x + threadIdx.x;
    if (i < n16) p[i] = uint4{0u, 0u, 0u, 0u};
}

__global__ void k_scatter(const int* __restrict__ ei, uint8_t* __restrict__ A) {
    int e = blockIdx.x * blockDim.x + threadIdx.x;
    if (e >= Ee) return;
    int s = ei[e], d = ei[Ee + e];
    A[(size_t)s * Nn + d] = FP8_ONE;
    A[(size_t)d * Nn + s] = FP8_ONE;
}

__global__ void k_mask_init(const uint32_t* __restrict__ A, uint32_t* __restrict__ M, int n4) {
    int i = blockIdx.x * blockDim.x + threadIdx.x;
    if (i >= n4) return;
    uint32_t a = A[i], m = 0;
    if (a & 0x000000ffu) m |= 0x00000001u;
    if (a & 0x0000ff00u) m |= 0x00000100u;
    if (a & 0x00ff0000u) m |= 0x00010000u;
    if (a & 0xff000000u) m |= 0x01000000u;
    M[i] = m;
}

__global__ __launch_bounds__(256) void k_gemm_bool(
    const uint8_t* __restrict__ Bin,
    const uint8_t* __restrict__ Adj,
    uint8_t*       __restrict__ Bout,
    uint8_t*       __restrict__ Mask,
    int hop)
{
    __shared__ __align__(16) uint8_t Smem[2 * 2 * TILE * TILE];
    const uint32_t lds_base = (uint32_t)(uintptr_t)(&Smem[0]);

    const int tid  = threadIdx.x;
    const int lane = tid & 31;
    const int wave = tid >> 5;
    const int h    = lane >> 4;
    const int lm   = lane & 15;
    const int s    = lm & 7;
    const int wm   = wave & 3;
    const int wn   = wave >> 2;
    const int mBase = blockIdx.x * TILE;
    const int nBase = blockIdx.y * TILE;

    auto stage = [&](int buf, int k0) {
        const uint32_t bb = lds_base + (uint32_t)buf * 32768u;
        #pragma unroll
        for (int i = 0; i < 4; ++i) {
            int idx = i * 256 + tid;
            int row = idx >> 3, cx = idx & 7;
            uint32_t sw = (uint32_t)(((cx ^ (row & 7)) << 4) + row * TILE);
            async_ld16(bb + sw,
                       Bin + (size_t)(mBase + row) * Nn + k0 + cx * 16);
            async_ld16(bb + 16384u + sw,
                       Adj + (size_t)(nBase + row) * Nn + k0 + cx * 16);
        }
    };

    v8f acc[2][4];
    #pragma unroll
    for (int mi = 0; mi < 2; ++mi)
        #pragma unroll
        for (int ni = 0; ni < 4; ++ni)
            acc[mi][ni] = v8f{0.f,0.f,0.f,0.f,0.f,0.f,0.f,0.f};

    stage(0, 0);
    for (int kt = 0; kt < Nn / TILE; ++kt) {
        wait_async_zero();
        __syncthreads();
        if (kt + 1 < Nn / TILE)
            stage((kt + 1) & 1, (kt + 1) * TILE);

        const uint8_t* Bs = &Smem[(kt & 1) * 32768];
        const uint8_t* As = Bs + 16384;

        AFrag af[2];
        #pragma unroll
        for (int mi = 0; mi < 2; ++mi) {
            int row = wm * 32 + mi * 16 + lm;
            #pragma unroll
            for (int p = 0; p < 8; ++p)
                af[mi].d[p] = *(const uint2*)(&Bs[row * TILE + (((2 * (p >> 1) + h) ^ s) << 4) + (p & 1) * 8]);
        }
        #pragma unroll
        for (int ni = 0; ni < 4; ++ni) {
            BFrag bf;
            int row = wn * 64 + ni * 16 + lm;
            #pragma unroll
            for (int q = 0; q < 4; ++q)
                bf.q[q] = *(const uint4*)(&As[row * TILE + (((2 * q + h) ^ s) << 4)]);
            #pragma unroll
            for (int mi = 0; mi < 2; ++mi)
                acc[mi][ni] = __builtin_amdgcn_wmma_f32_16x16x128_fp8_fp8(
                    af[mi].v, bf.v, (short)0, acc[mi][ni],
                     false,  false);
        }
    }

    #pragma unroll
    for (int mi = 0; mi < 2; ++mi)
        #pragma unroll
        for (int ni = 0; ni < 4; ++ni)
            #pragma unroll
            for (int r = 0; r < 8; ++r) {
                int m = mBase + wm * 32 + mi * 16 + r + 8 * h;
                int n = nBase + wn * 64 + ni * 16 + lm;
                size_t idx = (size_t)m * Nn + n;
                float c = acc[mi][ni][r];
                uint8_t b = (c > 0.5f) ? FP8_ONE : (uint8_t)0;
                Bout[idx] = b;
                if (b && Mask[idx] == 0) Mask[idx] = (uint8_t)hop;
            }
}

__global__ void k_compose(const float* __restrict__ emb,
                          const float* __restrict__ ru,
                          const uint8_t* __restrict__ M,
                          float* __restrict__ out)
{
    size_t base = ((size_t)blockIdx.x * blockDim.x + threadIdx.x) * 4;
    if (base >= (size_t)Nn * Nn) return;
    int i = (int)(base >> 12);
    int j = (int)(base & (Nn - 1));
    uint32_t m4 = *(const uint32_t*)(M + base);
    float4 r4 = *(const float4*)(ru + base);
    float rr[4] = {r4.x, r4.y, r4.z, r4.w};
    float v[4];
    #pragma unroll
    for (int t = 0; t < 4; ++t) {
        uint8_t m = (uint8_t)((m4 >> (8 * t)) & 0xffu);
        float val;
        if (i == j + t)      val = emb[0];
        else if (m)          val = emb[m];
        else                 val = (rr[t] > 0.01f) ? -__builtin_inff() : 0.0f;
        v[t] = val;
    }
    *(float4*)(out + base) = make_float4(v[0], v[1], v[2], v[3]);
}

extern "C" void kernel_launch(void* const* d_in, const int* in_sizes, int n_in,
                              void* d_out, int out_size, void* d_ws, size_t ws_size,
                              hipStream_t stream) {
    const float* emb   = (const float*)d_in[0];
    const float* randu = (const float*)d_in[1];
    const int*   eidx  = (const int*)d_in[2];
    float*       out   = (float*)d_out;

    const size_t MAT = (size_t)Nn * Nn;
    uint8_t* A  = (uint8_t*)d_ws;
    uint8_t* M  = A + MAT;
    uint8_t* B2 = A + 2 * MAT;
    uint8_t* B3 = A + 3 * MAT;

    {
        int n16 = (int)(2 * MAT / 16);
        k_zero<<<(n16 + 255) / 256, 256, 0, stream>>>((uint4*)A, n16);
    }
    k_scatter<<<(Ee + 255) / 256, 256, 0, stream>>>(eidx, A);
    {
        int n4 = (int)(MAT / 4);
        k_mask_init<<<(n4 + 255) / 256, 256, 0, stream>>>((const uint32_t*)A,
                                                          (uint32_t*)M, n4);
    }
    dim3 grid(Nn / TILE, Nn / TILE);
    k_gemm_bool<<<grid, 256, 0, stream>>>(A,  A, B2, M, 2);
    k_gemm_bool<<<grid, 256, 0, stream>>>(B2, A, B3, M, 3);
    k_gemm_bool<<<grid, 256, 0, stream>>>(B3, A, B2, M, 4);
    k_compose<<<(int)(MAT / 4 + 255) / 256, 256, 0, stream>>>(emb, randu, M, out);
}
